// MHSA_V1_16389595202178
// MI455X (gfx1250) — matrix operand order repaired and checked by reading, not compiled, not run on hardware
//
#include <hip/hip_runtime.h>
#include <hip/hip_bf16.h>

// ---------------------------------------------------------------------------
// MHSA with relative position bias for MI455X (gfx1250), bf16 WMMA pipeline.
// B=8, C=512, H=8, D=64, N=1024 (W=Hh=32).
// ---------------------------------------------------------------------------

#define BATCH 8
#define CH    512
#define HEADS 8
#define HD    64
#define NTOK  1024
#define WW    32
#define HH    32

typedef __bf16 bf16;
typedef __attribute__((ext_vector_type(16))) __bf16 bfx16;
typedef __attribute__((ext_vector_type(8)))  __bf16 bfx8;
typedef __attribute__((ext_vector_type(4)))  __bf16 bfx4;
typedef __attribute__((ext_vector_type(8)))  float  fx8;

static __device__ __forceinline__ bf16 f2bf(float f) { return (bf16)f; }

static __device__ __forceinline__ fx8 wmma_bf16(bfx16 a, bfx16 b, fx8 c) {
  // D = A(16x32) * B(32x16) + C, f32 accumulate
  return __builtin_amdgcn_wmma_f32_16x16x32_bf16(
      /*neg_a=*/false, a, /*neg_b=*/false, b,
      /*c_mod=*/(short)0, c, /*reuse_a=*/false, /*reuse_b=*/false);
}

// A-operand k order used here (B matches): lane m = lane&15, half = lane>>4.
// element j holds K = 16*half + j  -> two contiguous 16B runs.
// Works for global or LDS pointers (two b128 loads).
static __device__ __forceinline__ bfx16 load_a16(const bf16* rowp, int half) {
  bfx8 lo = *(const bfx8*)(rowp + 16 * half);
  bfx8 hi = *(const bfx8*)(rowp + 16 * half + 8);
  bfx16 a;
#pragma unroll
  for (int i = 0; i < 8; ++i) { a[i] = lo[i]; a[8 + i] = hi[i]; }
  return a;
}

// CDNA5 async copy: per-lane 16B global -> LDS, tracked with ASYNCcnt.
// lds_off = wave-relative LDS byte offset (low 32 bits of LDS pointer).
static __device__ __forceinline__ void async_g2l_b128(unsigned lds_off, const bf16* g) {
  asm volatile("global_load_async_to_lds_b128 %0, %1, off"
               :: "v"(lds_off), "v"(g) : "memory");
}
static __device__ __forceinline__ void wait_asynccnt0() {
  asm volatile("s_wait_asynccnt 0x0" ::: "memory");
}

// ---------------------------------------------------------------------------
// Kernel 1: convert Wq/Wk/Wv (f32 row-major [o][c]) to bf16.
// ---------------------------------------------------------------------------
__global__ void k_convert_w(const float* __restrict__ Wq,
                            const float* __restrict__ Wk,
                            const float* __restrict__ Wv,
                            bf16* __restrict__ Wb) {
  int i = blockIdx.x * 256 + threadIdx.x;         // 0 .. 3*CH*CH-1
  const int sz = CH * CH;
  const float* src = (i < sz) ? Wq : (i < 2 * sz ? Wk : Wv);
  Wb[i] = f2bf(src[i % sz]);
}

// ---------------------------------------------------------------------------
// Kernel 2: transpose x [B,C,N] f32 -> x_t [B,N,C] bf16  (LDS 32x32 tiles).
// ---------------------------------------------------------------------------
__global__ void k_transpose_x(const float* __restrict__ x, bf16* __restrict__ xt) {
  __shared__ float tile[32][33];
  int b = blockIdx.z, c0 = blockIdx.y * 32, n0 = blockIdx.x * 32;
  int tx = threadIdx.x, ty = threadIdx.y;          // 32 x 8
  const float* xp = x + ((size_t)b * CH + c0) * NTOK + n0;
#pragma unroll
  for (int k = 0; k < 4; ++k) tile[ty + 8 * k][tx] = xp[(size_t)(ty + 8 * k) * NTOK + tx];
  __syncthreads();
  bf16* op = xt + ((size_t)b * NTOK + n0) * CH + c0;
#pragma unroll
  for (int k = 0; k < 4; ++k) op[(size_t)(ty + 8 * k) * CH + tx] = f2bf(tile[tx][ty + 8 * k]);
}

// ---------------------------------------------------------------------------
// Kernel 3: rel_t [H,N,D] bf16 = rel_h[h,d,hh] + rel_w[h,d,w], n = w*HH + hh.
// ---------------------------------------------------------------------------
__global__ void k_rel(const float* __restrict__ rel_h,
                      const float* __restrict__ rel_w,
                      bf16* __restrict__ relt) {
  int i = blockIdx.x * 256 + threadIdx.x;          // h*N*D + n*D + d
  int d = i & (HD - 1);
  int n = (i / HD) & (NTOK - 1);
  int h = i / (HD * NTOK);
  int w = n / HH, hh = n & (HH - 1);
  float v = rel_h[(h * HD + d) * HH + hh] + rel_w[(h * HD + d) * WW + w];
  relt[i] = f2bf(v);
}

// ---------------------------------------------------------------------------
// Kernel 4: QKV projection GEMM, block-cooperative 64(o) x 64(n) macro-tile.
// The 4 waves share a double-buffered LDS B-chunk (64 tok x 32 c = 4KB),
// filled with GLOBAL_LOAD_ASYNC_TO_LDS_B128 (ASYNCcnt) so the copy of chunk
// k+1 overlaps the WMMAs of chunk k.  A comes per-wave from global (L2).
//   q,k -> token-major [B,N,C] bf16 ; v -> dim-major [B,C,N] bf16.
// ---------------------------------------------------------------------------
__global__ __launch_bounds__(128) void k_qkv(
    const bf16* __restrict__ Wb, const bf16* __restrict__ xt,
    const float* __restrict__ bq, const float* __restrict__ bk,
    const float* __restrict__ bv,
    bf16* __restrict__ qt, bf16* __restrict__ kt, bf16* __restrict__ vc) {
  __shared__ bf16 Bbuf[2][64 * 32];                // 2 x 4KB
  int tid = threadIdx.x;
  int wvid = tid >> 5, lane = tid & 31;
  int m_l = lane & 15, half = lane >> 4;

  int blk = blockIdx.x;                 // ((proj*BATCH + b)*8 + om)*16 + ng
  int ng = blk & 15;  int t = blk >> 4;
  int om = t & 7;     t >>= 3;
  int b  = t & 7;     int proj = t >> 3;

  const bf16* arow = Wb + ((size_t)proj * CH + (om * 64 + wvid * 16 + m_l)) * CH;
  const bf16* bsrc = xt + ((size_t)b * NTOK + ng * 64) * CH;   // 64 token rows

  // staging split: thread covers 32B -> token = tid>>1, 16 elements at (tid&1)*16
  int stok = tid >> 1;
  int scof = (tid & 1) * 16;
  const bf16* gstage = bsrc + (size_t)stok * CH + scof;
  unsigned lbase = (unsigned)(uintptr_t)(&Bbuf[0][stok * 32 + scof]);
  const unsigned lstride = (unsigned)(64 * 32 * sizeof(bf16));  // buffer stride

  fx8 acc0 = {}, acc1 = {}, acc2 = {}, acc3 = {};

  // prologue: stage chunk 0 into buffer 0
  async_g2l_b128(lbase, gstage);
  async_g2l_b128(lbase + 16, gstage + 8);

#pragma unroll 2
  for (int ic = 0; ic < 16; ++ic) {
    int kc = ic * 32;
    int cur = ic & 1;
    wait_asynccnt0();
    __syncthreads();                       // buf[cur] fully populated & visible
    if (ic + 1 < 16) {                     // overlap next copy with compute
      unsigned ldst = lbase + (unsigned)((ic + 1) & 1) * lstride;
      const bf16* g = gstage + (kc + 32);
      async_g2l_b128(ldst, g);
      async_g2l_b128(ldst + 16, g + 8);
    }
    bfx16 a = load_a16(arow + kc, half);
    const bf16* bb = &Bbuf[cur][m_l * 32 + 16 * half];
    bfx16 b0 = *(const bfx16*)(bb);
    bfx16 b1 = *(const bfx16*)(bb + 16 * 32);
    bfx16 b2 = *(const bfx16*)(bb + 32 * 32);
    bfx16 b3 = *(const bfx16*)(bb + 48 * 32);
    acc0 = wmma_bf16(a, b0, acc0);
    acc1 = wmma_bf16(a, b1, acc1);
    acc2 = wmma_bf16(a, b2, acc2);
    acc3 = wmma_bf16(a, b3, acc3);
    __syncthreads();                       // reads done before buf[cur] reuse
  }

  const float* bias = (proj == 0) ? bq : (proj == 1 ? bk : bv);
  int o0 = om * 64 + wvid * 16 + 8 * half;  // lane's first output channel
  float bi[8];
#pragma unroll
  for (int r = 0; r < 8; ++r) bi[r] = bias[o0 + r];

  fx8 accs[4] = {acc0, acc1, acc2, acc3};
#pragma unroll
  for (int q4 = 0; q4 < 4; ++q4) {
    int n = ng * 64 + q4 * 16 + m_l;        // lane's token column
    if (proj < 2) {
      bfx8 pk;
#pragma unroll
      for (int r = 0; r < 8; ++r) pk[r] = f2bf(accs[q4][r] + bi[r]);
      bf16* dst = (proj == 0 ? qt : kt) + ((size_t)b * NTOK + n) * CH + o0;
      *(bfx8*)dst = pk;                     // 16B contiguous store
    } else {
#pragma unroll
      for (int r = 0; r < 8; ++r)
        vc[((size_t)b * CH + (o0 + r)) * NTOK + n] = f2bf(accs[q4][r] + bi[r]);
    }
  }
}

// ---------------------------------------------------------------------------
// Kernel 5: fused attention. One workgroup (4 waves) per (b, h, 16-row mtile).
//   S[m,n] = q_m . k_n + rel_m . q_n   (K = 64 over 2 chunks, A hoisted)
//   row softmax in LDS (float4 passes), normalized P written ONCE as bf16;
//   O_t[m][d] = sum_n P[m,n] v[d,n].
// ---------------------------------------------------------------------------
__global__ __launch_bounds__(128) void k_attn(
    const bf16* __restrict__ qt, const bf16* __restrict__ kt,
    const bf16* __restrict__ vc, const bf16* __restrict__ relt,
    float* __restrict__ out) {
  __shared__ float S[16 * NTOK];       // 64 KB scores / exp
  __shared__ bf16  P[16 * NTOK];       // 32 KB normalized probs (bf16)
  __shared__ float red[16][8];
  __shared__ float rmax[16];
  __shared__ float rinv[16];

  int tid = threadIdx.x;
  int wv = tid >> 5, lane = tid & 31;
  int m_l = lane & 15, half = lane >> 4;

  int blk = blockIdx.x;                // ((b*HEADS)+h)*64 + mt
  int mt = blk & 63; int bh = blk >> 6;
  int h = bh & 7;    int b = bh >> 3;

  // ---- pass 1: score tiles (independent iterations, unrolled x2) ----------
  int mglob = mt * 16 + m_l;
  const bf16* qrow = qt + ((size_t)b * NTOK + mglob) * CH + h * HD;
  const bf16* rrow = relt + ((size_t)h * NTOK + mglob) * HD;
  bfx16 aq0 = load_a16(qrow + 0,  half);
  bfx16 aq1 = load_a16(qrow + 32, half);
  bfx16 ar0 = load_a16(rrow + 0,  half);
  bfx16 ar1 = load_a16(rrow + 32, half);

#pragma unroll 2
  for (int t2 = 0; t2 < 16; t2 += 2) {
    int n0 = wv * 256 + t2 * 16;
    const bf16* kr0 = kt + ((size_t)b * NTOK + (n0 + m_l)) * CH + h * HD + 16 * half;
    const bf16* qr0 = qt + ((size_t)b * NTOK + (n0 + m_l)) * CH + h * HD + 16 * half;
    const bf16* kr1 = kr0 + 16 * CH;
    const bf16* qr1 = qr0 + 16 * CH;
    bfx16 bk0 = *(const bfx16*)(kr0);
    bfx16 bk1 = *(const bfx16*)(kr0 + 32);
    bfx16 bq0 = *(const bfx16*)(qr0);
    bfx16 bq1 = *(const bfx16*)(qr0 + 32);
    bfx16 ck0 = *(const bfx16*)(kr1);
    bfx16 ck1 = *(const bfx16*)(kr1 + 32);
    bfx16 cq0 = *(const bfx16*)(qr1);
    bfx16 cq1 = *(const bfx16*)(qr1 + 32);
    fx8 accA = {}, accB = {};
    accA = wmma_bf16(aq0, bk0, accA);
    accB = wmma_bf16(aq0, ck0, accB);
    accA = wmma_bf16(aq1, bk1, accA);
    accB = wmma_bf16(aq1, ck1, accB);
    accA = wmma_bf16(ar0, bq0, accA);
    accB = wmma_bf16(ar0, cq0, accB);
    accA = wmma_bf16(ar1, bq1, accA);
    accB = wmma_bf16(ar1, cq1, accB);
    // C/D layout: lane holds column n0+m_l, vgpr r -> row r + 8*half
#pragma unroll
    for (int r = 0; r < 8; ++r) {
      S[(r + 8 * half) * NTOK + n0 + m_l]      = accA[r];
      S[(r + 8 * half) * NTOK + n0 + 16 + m_l] = accB[r];
    }
  }
  __syncthreads();

  // ---- softmax (16 rows x 1024 cols; 8 threads per row; float4 passes) ----
  int row = tid >> 3;
  int seg = tid & 7;
  float4* srow4 = (float4*)(S + row * NTOK + seg * 128);
  float mx = -3.0e38f;
#pragma unroll 8
  for (int i = 0; i < 32; ++i) {
    float4 v = srow4[i];
    mx = fmaxf(mx, fmaxf(fmaxf(v.x, v.y), fmaxf(v.z, v.w)));
  }
  red[row][seg] = mx;
  __syncthreads();
  if (tid < 16) {
    float m2 = red[tid][0];
#pragma unroll
    for (int i = 1; i < 8; ++i) m2 = fmaxf(m2, red[tid][i]);
    rmax[tid] = m2;
  }
  __syncthreads();
  float rm = rmax[row];
  float s = 0.f;
#pragma unroll 8
  for (int i = 0; i < 32; ++i) {
    float4 v = srow4[i];
    v.x = __expf(v.x - rm);
    v.y = __expf(v.y - rm);
    v.z = __expf(v.z - rm);
    v.w = __expf(v.w - rm);
    s += (v.x + v.y) + (v.z + v.w);
    srow4[i] = v;
  }
  red[row][seg] = s;
  __syncthreads();
  if (tid < 16) {
    float s2 = 0.f;
#pragma unroll
    for (int i = 0; i < 8; ++i) s2 += red[tid][i];
    rinv[tid] = 1.0f / s2;
  }
  __syncthreads();
  // normalize + convert to bf16 once
  float nv = rinv[row];
  bf16* prow_w = P + row * NTOK + seg * 128;
#pragma unroll 8
  for (int i = 0; i < 32; ++i) {
    float4 v = srow4[i];
    bfx4 pk;
    pk[0] = f2bf(v.x * nv);
    pk[1] = f2bf(v.y * nv);
    pk[2] = f2bf(v.z * nv);
    pk[3] = f2bf(v.w * nv);
    *(bfx4*)(prow_w + i * 4) = pk;   // 8B LDS store
  }
  __syncthreads();

  // ---- pass 2: O_t = P * V^T, one 16-wide d-tile per wave -----------------
  const bf16* prow = P + (size_t)m_l * NTOK;
  const bf16* vrow = vc + ((size_t)b * CH + (h * HD + wv * 16 + m_l)) * NTOK + 16 * half;
  fx8 o0 = {}, o1 = {};
  bfx16 vb0 = *(const bfx16*)(vrow);
  bfx16 vb1 = *(const bfx16*)(vrow + 32);
#pragma unroll
  for (int kn = 0; kn < 32; kn += 2) {
    bfx16 vn0 = vb0, vn1 = vb1;
    if (kn + 2 < 32) {
      vn0 = *(const bfx16*)(vrow + (kn + 2) * 32);
      vn1 = *(const bfx16*)(vrow + (kn + 3) * 32);
    }
    bfx16 a0 = load_a16(prow + kn * 32, half);
    bfx16 a1 = load_a16(prow + (kn + 1) * 32, half);
    o0 = wmma_bf16(a0, vb0, o0);
    o1 = wmma_bf16(a1, vb1, o1);
    vb0 = vn0; vb1 = vn1;
  }
  fx8 oacc = o0 + o1;
  // D layout: lane col = d local (lane&15), vgpr r -> m = r + 8*half
  float* obase = out + ((size_t)b * CH + (h * HD + wv * 16 + m_l)) * NTOK + mt * 16 + 8 * half;
  *(float4*)(obase)     = make_float4(oacc[0], oacc[1], oacc[2], oacc[3]);
  *(float4*)(obase + 4) = make_float4(oacc[4], oacc[5], oacc[6], oacc[7]);
}

// ---------------------------------------------------------------------------
// Launch
// ---------------------------------------------------------------------------
extern "C" void kernel_launch(void* const* d_in, const int* in_sizes, int n_in,
                              void* d_out, int out_size, void* d_ws, size_t ws_size,
                              hipStream_t stream) {
  const float* x     = (const float*)d_in[0];
  const float* Wq    = (const float*)d_in[1];
  const float* bq    = (const float*)d_in[2];
  const float* Wk    = (const float*)d_in[3];
  const float* bk    = (const float*)d_in[4];
  const float* Wv    = (const float*)d_in[5];
  const float* bv    = (const float*)d_in[6];
  const float* rel_h = (const float*)d_in[7];
  const float* rel_w = (const float*)d_in[8];
  float* out = (float*)d_out;

  // workspace layout (bytes); total ~34.5 MB
  char* ws = (char*)d_ws;
  bf16* Wb   = (bf16*)(ws + 0);                         // 3*512*512*2   = 1.50 MB
  bf16* xt   = (bf16*)(ws + 1572864);                   // 8*1024*512*2  = 8 MB
  bf16* relt = (bf16*)(ws + 9961472);                   // 8*1024*64*2   = 1 MB
  bf16* qt   = (bf16*)(ws + 11010048);                  // 8 MB
  bf16* kt   = (bf16*)(ws + 19398656);                  // 8 MB
  bf16* vc   = (bf16*)(ws + 27787264);                  // 8 MB

  k_convert_w<<<(3 * CH * CH) / 256, 256, 0, stream>>>(Wq, Wk, Wv, Wb);
  k_transpose_x<<<dim3(NTOK / 32, CH / 32, BATCH), dim3(32, 8, 1), 0, stream>>>(x, xt);
  k_rel<<<(HEADS * NTOK * HD) / 256, 256, 0, stream>>>(rel_h, rel_w, relt);

  // blocks: 3 proj * 8 batch * 8 o-macros * 16 n-groups
  k_qkv<<<3 * BATCH * 8 * 16, 128, 0, stream>>>(Wb, xt, bq, bk, bv, qt, kt, vc);

  // one block per (b, h, mtile)
  k_attn<<<BATCH * HEADS * (NTOK / 16), 128, 0, stream>>>(qt, kt, vc, relt, out);
}
